// Attention_Block_63977832841464
// MI455X (gfx1250) — matrix operand order repaired and checked by reading, not compiled, not run on hardware
//
#include <hip/hip_runtime.h>
#include <hip/hip_bf16.h>

typedef __bf16 bf16_t;
typedef __attribute__((ext_vector_type(16))) __bf16 v16bf;
typedef __attribute__((ext_vector_type(8))) __bf16 v8bf;
typedef __attribute__((ext_vector_type(8))) float v8f;
typedef __attribute__((ext_vector_type(4))) unsigned int v4u;
typedef __attribute__((ext_vector_type(8))) int v8i;
typedef __attribute__((ext_vector_type(4))) int v4i;

#define D_MODEL 512
#define HEADS 8
#define HEAD_DIM 64
#define BATCH 4
#define SEQ 2048
#define NROWS (BATCH * SEQ)           /* 8192 */
#define NBH (BATCH * HEADS)           /* 32 */
#define HEAD_ELEMS (SEQ * HEAD_DIM)   /* 131072 */
#define W_ELEMS (D_MODEL * D_MODEL)   /* 262144 */

// ---------------- TDM: 2-D tile load Global -> LDS ----------------
// D# packing per CDNA5 ISA (2-D tensor, groups 2/3 zero).
__device__ __forceinline__ unsigned lds_off(const void* p) {
  return (unsigned)(uintptr_t)p;  // low 32 bits of shared-aperture flat addr
}
__device__ __forceinline__ void tdm_load_2d(unsigned lds_byte, const void* gptr,
                                            unsigned data_size_code,
                                            unsigned row_elems, unsigned rows,
                                            unsigned row_stride, unsigned tensor_rows) {
  const unsigned long long ga = (unsigned long long)(uintptr_t)gptr;
  v4u g0;
  g0[0] = 1u;                                                  // count=1, user mode
  g0[1] = lds_byte;                                            // lds_addr
  g0[2] = (unsigned)(ga & 0xffffffffu);                        // global_addr[31:0]
  g0[3] = (unsigned)((ga >> 32) & 0x01ffffffu) | (2u << 30);   // addr[56:32] | type=2
  v8i g1;
  g1[0] = (int)(data_size_code << 16);                         // wg_mask=0 | data_size
  g1[1] = (int)(row_elems << 16);                              // tensor_dim0[15:0]
  g1[2] = (int)((row_elems >> 16) | (tensor_rows << 16));      // dim0 hi | dim1 lo
  g1[3] = (int)((tensor_rows >> 16) | (row_elems << 16));      // dim1 hi | tile_dim0
  g1[4] = (int)(rows & 0xffffu);                               // tile_dim1
  g1[5] = (int)row_stride;                                     // tensor_dim0_stride lo
  g1[6] = 0;
  g1[7] = 0;
  const v4i z = {0, 0, 0, 0};
#if defined(__clang_major__) && (__clang_major__ >= 23)
  const v8i z8 = {0, 0, 0, 0, 0, 0, 0, 0};
  __builtin_amdgcn_tensor_load_to_lds(g0, g1, z, z, z8, 0);
#else
  __builtin_amdgcn_tensor_load_to_lds(g0, g1, z, z, 0);
#endif
}
__device__ __forceinline__ void tdm_wait() { __builtin_amdgcn_s_wait_tensorcnt(0); }

// ---------------- WMMA fragment helpers (CDNA5 wave32 layouts) ----------------
__device__ __forceinline__ v16bf load_a_bf16(const bf16_t* base, int lda, int lane) {
  const int row = lane & 15, kg = lane >> 4;
  const bf16_t* p = base + (size_t)row * lda + kg * 16;
  v16bf a;
#pragma unroll
  for (int e = 0; e < 16; ++e) a[e] = p[e];
  return a;
}
// B (32x16): lane holds col = lane&15; K = (lane>>4)*16 + e; k-contiguous source
__device__ __forceinline__ v16bf load_b_kcontig_bf16(const bf16_t* base, int ldb, int lane) {
  const int col = lane & 15, kg = lane >> 4;
  return *(const v16bf*)(base + (size_t)col * ldb + kg * 16);  // 32B aligned
}
__device__ __forceinline__ v16bf load_b_rowmajor_bf16(const bf16_t* base, int ldb, int lane) {
  const int col = lane & 15, kg = lane >> 4;
  const bf16_t* p = base + (size_t)(kg * 16) * ldb + col;
  v16bf b;
#pragma unroll
  for (int e = 0; e < 16; ++e) b[e] = p[(size_t)e * ldb];
  return b;
}
__device__ __forceinline__ v8f wmma_bf16(v16bf a, v16bf b, v8f c) {
  return __builtin_amdgcn_wmma_f32_16x16x32_bf16(false, a, false, b, (short)0, c,
                                                 false, false);
}

// ---------------- Kernel 0: one-shot fp32 -> bf16 conversion ----------------
// Bandwidth-bound pre-pass so the GEMM inner loops carry zero converts.
__global__ __launch_bounds__(256) void cvt_bf16_kernel(
    const float* __restrict__ src, bf16_t* __restrict__ dst, int n) {
  const int i = (blockIdx.x * 256 + threadIdx.x) * 8;
  if (i + 8 <= n) {
    const float4 f0 = *(const float4*)(src + i);
    const float4 f1 = *(const float4*)(src + i + 4);
    v8bf o;
    o[0] = (bf16_t)f0.x; o[1] = (bf16_t)f0.y; o[2] = (bf16_t)f0.z; o[3] = (bf16_t)f0.w;
    o[4] = (bf16_t)f1.x; o[5] = (bf16_t)f1.y; o[6] = (bf16_t)f1.z; o[7] = (bf16_t)f1.w;
    *(v8bf*)(dst + i) = o;
  }
}

// ---------------- Kernel 1: fused QKV projection (all-bf16) ----------------
// grid=(NROWS/16, 4), block=256 (8 waves). Shared x-tile staged once via TDM.
__global__ __launch_bounds__(256) void qkv_proj_kernel(
    const bf16_t* __restrict__ xb, const bf16_t* __restrict__ Wq,
    const bf16_t* __restrict__ Wk, const bf16_t* __restrict__ Wv,
    bf16_t* __restrict__ Qb, bf16_t* __restrict__ Kb, bf16_t* __restrict__ Vb) {
  __shared__ bf16_t xs[16 * D_MODEL];  // 16 KB
  const int lane = threadIdx.x & 31;
  const int wave = threadIdx.x >> 5;
  const int row0 = blockIdx.x * 16;
  const int col0 = blockIdx.y * 128 + wave * 16;
  if (wave == 0) {
    tdm_load_2d(lds_off(xs), xb + (size_t)row0 * D_MODEL, /*2B*/1,
                D_MODEL, 16, D_MODEL, NROWS);
    tdm_wait();
  }
  __syncthreads();
  v8f cq = {}, ck = {}, cv = {};
  for (int kk = 0; kk < D_MODEL; kk += 32) {
    v16bf a = load_a_bf16(xs + kk, D_MODEL, lane);
    v16bf bq = load_b_kcontig_bf16(Wq + (size_t)col0 * D_MODEL + kk, D_MODEL, lane);
    v16bf bk = load_b_kcontig_bf16(Wk + (size_t)col0 * D_MODEL + kk, D_MODEL, lane);
    v16bf bv = load_b_kcontig_bf16(Wv + (size_t)col0 * D_MODEL + kk, D_MODEL, lane);
    cq = wmma_bf16(a, bq, cq);
    ck = wmma_bf16(a, bk, ck);
    cv = wmma_bf16(a, bv, cv);
  }
  const int col = lane & 15, kg = lane >> 4;
#pragma unroll
  for (int r = 0; r < 8; ++r) {
    const size_t off = (size_t)(row0 + kg * 8 + r) * D_MODEL + col0 + col;
    Qb[off] = (bf16_t)cq[r];
    Kb[off] = (bf16_t)ck[r];
    Vb[off] = (bf16_t)cv[r];
  }
}

// ---------------- Kernel 2: column softmax denominators ----------------
// rcp[bh,k] = 1 / sum_q exp(q.k/8); grid=(SEQ/128, NBH). Shared Q tiles:
// double-buffered TDM stream.
__global__ __launch_bounds__(256) void colsum_kernel(
    const bf16_t* __restrict__ Qb, const bf16_t* __restrict__ Kb,
    float* __restrict__ rcp) {
  __shared__ bf16_t qs[2][16 * HEAD_DIM];  // 2 x 2 KB
  const int lane = threadIdx.x & 31;
  const int wave = threadIdx.x >> 5;
  const int bh = blockIdx.y;
  const int kbase = blockIdx.x * 128 + wave * 16;
  const bf16_t* Qh = Qb + (size_t)bh * HEAD_ELEMS;
  const bf16_t* Kh = Kb + (size_t)bh * HEAD_ELEMS;
  // K^T B-fragments: loop-invariant
  const v16bf b0 = load_b_kcontig_bf16(Kh + (size_t)kbase * HEAD_DIM + 0, HEAD_DIM, lane);
  const v16bf b1 = load_b_kcontig_bf16(Kh + (size_t)kbase * HEAD_DIM + 32, HEAD_DIM, lane);
  if (wave == 0) {
    tdm_load_2d(lds_off(qs[0]), Qh, 1, HEAD_DIM, 16, HEAD_DIM, SEQ);
    tdm_wait();
  }
  __syncthreads();
  float s = 0.f;
  for (int q0 = 0; q0 < SEQ; q0 += 16) {
    const int cur = (q0 >> 4) & 1;
    if (wave == 0 && q0 + 16 < SEQ)
      tdm_load_2d(lds_off(qs[cur ^ 1]), Qh + (size_t)(q0 + 16) * HEAD_DIM, 1,
                  HEAD_DIM, 16, HEAD_DIM, SEQ);
    v8f c = {};
    v16bf a0 = load_a_bf16(qs[cur] + 0, HEAD_DIM, lane);
    v16bf a1 = load_a_bf16(qs[cur] + 32, HEAD_DIM, lane);
    c = wmma_bf16(a0, b0, c);
    c = wmma_bf16(a1, b1, c);
#pragma unroll
    for (int r = 0; r < 8; ++r) s += __expf(c[r] * 0.125f);
    if (wave == 0) tdm_wait();
    __syncthreads();
  }
  s += __shfl_xor(s, 16, 32);  // combine the two row-halves of each column
  if (lane < 16) rcp[(size_t)bh * SEQ + kbase + lane] = 1.0f / s;
}

// ---------------- Kernel 3: out = exp(S/8) * diag(rcp) @ V ----------------
// grid=(SEQ/128, NBH); wave owns 16 q-rows x 64 d. Shared K/V tiles:
// double-buffered TDM stream.
__global__ __launch_bounds__(256) void attn_out_kernel(
    const bf16_t* __restrict__ Qb, const bf16_t* __restrict__ Kb,
    const bf16_t* __restrict__ Vb, const float* __restrict__ rcp,
    bf16_t* __restrict__ Ab) {
  __shared__ bf16_t ks[2][32 * HEAD_DIM];  // 2 x 4 KB
  __shared__ bf16_t vs[2][32 * HEAD_DIM];  // 2 x 4 KB
  __shared__ bf16_t pbuf[8][16 * 32];      // per-wave P staging (C->A relayout)
  const int lane = threadIdx.x & 31;
  const int wave = threadIdx.x >> 5;
  const int bh = blockIdx.y;
  const int qbase = blockIdx.x * 128 + wave * 16;
  const bf16_t* Qh = Qb + (size_t)bh * HEAD_ELEMS;
  const bf16_t* Kh = Kb + (size_t)bh * HEAD_ELEMS;
  const bf16_t* Vh = Vb + (size_t)bh * HEAD_ELEMS;
  const float* rch = rcp + (size_t)bh * SEQ;
  const int col = lane & 15, kg = lane >> 4;
  bf16_t* pl = pbuf[wave];

  // Q A-fragments: loop-invariant
  const v16bf aq0 = load_a_bf16(Qh + (size_t)qbase * HEAD_DIM + 0, HEAD_DIM, lane);
  const v16bf aq1 = load_a_bf16(Qh + (size_t)qbase * HEAD_DIM + 32, HEAD_DIM, lane);
  v8f acc0 = {}, acc1 = {}, acc2 = {}, acc3 = {};

  if (wave == 0) {
    tdm_load_2d(lds_off(ks[0]), Kh, 1, HEAD_DIM, 32, HEAD_DIM, SEQ);
    tdm_load_2d(lds_off(vs[0]), Vh, 1, HEAD_DIM, 32, HEAD_DIM, SEQ);
    tdm_wait();
  }
  __syncthreads();

  for (int k0 = 0; k0 < SEQ; k0 += 32) {
    const int cur = (k0 >> 5) & 1;
    if (wave == 0 && k0 + 32 < SEQ) {  // async-stream next K/V tiles
      tdm_load_2d(lds_off(ks[cur ^ 1]), Kh + (size_t)(k0 + 32) * HEAD_DIM, 1,
                  HEAD_DIM, 32, HEAD_DIM, SEQ);
      tdm_load_2d(lds_off(vs[cur ^ 1]), Vh + (size_t)(k0 + 32) * HEAD_DIM, 1,
                  HEAD_DIM, 32, HEAD_DIM, SEQ);
    }
    const bf16_t* kt = ks[cur];
    const bf16_t* vt = vs[cur];
    // S tile: 16q x 32k as two 16x16 C-fragments
    v8f s0 = {}, s1 = {};
    s0 = wmma_bf16(aq0, load_b_kcontig_bf16(kt + 0, HEAD_DIM, lane), s0);
    s0 = wmma_bf16(aq1, load_b_kcontig_bf16(kt + 32, HEAD_DIM, lane), s0);
    s1 = wmma_bf16(aq0, load_b_kcontig_bf16(kt + 16 * HEAD_DIM + 0, HEAD_DIM, lane), s1);
    s1 = wmma_bf16(aq1, load_b_kcontig_bf16(kt + 16 * HEAD_DIM + 32, HEAD_DIM, lane), s1);

    const float r0 = rch[k0 + col];
    const float r1 = rch[k0 + 16 + col];
#pragma unroll
    for (int r = 0; r < 8; ++r) {  // P = exp(S/8)*rcp[k]; C-layout -> row-major LDS
      const int lrow = r + kg * 8;
      pl[lrow * 32 + col]      = (bf16_t)(__expf(s0[r] * 0.125f) * r0);
      pl[lrow * 32 + 16 + col] = (bf16_t)(__expf(s1[r] * 0.125f) * r1);
    }
    // same-wave DS ops complete in order: reload as A-fragment without barrier
    const v16bf ap = load_a_bf16(pl, 32, lane);
    acc0 = wmma_bf16(ap, load_b_rowmajor_bf16(vt + 0, HEAD_DIM, lane), acc0);
    acc1 = wmma_bf16(ap, load_b_rowmajor_bf16(vt + 16, HEAD_DIM, lane), acc1);
    acc2 = wmma_bf16(ap, load_b_rowmajor_bf16(vt + 32, HEAD_DIM, lane), acc2);
    acc3 = wmma_bf16(ap, load_b_rowmajor_bf16(vt + 48, HEAD_DIM, lane), acc3);
    if (wave == 0) tdm_wait();  // next tiles landed
    __syncthreads();            // publish double buffer
  }
  bf16_t* Ah = Ab + (size_t)bh * HEAD_ELEMS;  // head layout == final flat layout
#pragma unroll
  for (int r = 0; r < 8; ++r) {
    const size_t row = (size_t)(qbase + kg * 8 + r) * HEAD_DIM;
    Ah[row + 0 + col]  = (bf16_t)acc0[r];
    Ah[row + 16 + col] = (bf16_t)acc1[r];
    Ah[row + 32 + col] = (bf16_t)acc2[r];
    Ah[row + 48 + col] = (bf16_t)acc3[r];
  }
}

// ---------------- Kernel 4: y = attn @ Wo.T + bo (fp32 out) ----------------
__global__ __launch_bounds__(256) void out_proj_kernel(
    const bf16_t* __restrict__ Ab, const bf16_t* __restrict__ Wo,
    const float* __restrict__ bo, float* __restrict__ out) {
  __shared__ bf16_t as[16 * D_MODEL];  // 16 KB
  const int lane = threadIdx.x & 31;
  const int wave = threadIdx.x >> 5;
  const int row0 = blockIdx.x * 16;
  const int col0 = blockIdx.y * 128 + wave * 16;
  if (wave == 0) {
    tdm_load_2d(lds_off(as), Ab + (size_t)row0 * D_MODEL, 1,
                D_MODEL, 16, D_MODEL, NROWS);
    tdm_wait();
  }
  __syncthreads();
  v8f c = {};
  for (int kk = 0; kk < D_MODEL; kk += 32) {
    v16bf a = load_a_bf16(as + kk, D_MODEL, lane);
    v16bf b = load_b_kcontig_bf16(Wo + (size_t)col0 * D_MODEL + kk, D_MODEL, lane);
    c = wmma_bf16(a, b, c);
  }
  const int col = lane & 15, kg = lane >> 4;
  const float bias = bo[col0 + col];
#pragma unroll
  for (int r = 0; r < 8; ++r)
    out[(size_t)(row0 + kg * 8 + r) * D_MODEL + col0 + col] = c[r] + bias;
}

// ---------------- host launcher ----------------
extern "C" void kernel_launch(void* const* d_in, const int* in_sizes, int n_in,
                              void* d_out, int out_size, void* d_ws, size_t ws_size,
                              hipStream_t stream) {
  const float* x  = (const float*)d_in[0];
  const float* Wq = (const float*)d_in[1];
  const float* Wk = (const float*)d_in[2];
  const float* Wv = (const float*)d_in[3];
  const float* Wo = (const float*)d_in[4];
  const float* bo = (const float*)d_in[5];
  float* out = (float*)d_out;

  const size_t elems = (size_t)NROWS * D_MODEL;  // 4,194,304
  bf16_t* Qb = (bf16_t*)d_ws;
  bf16_t* Kb  = Qb + elems;
  bf16_t* Vb  = Kb + elems;
  bf16_t* Ab  = Vb + elems;
  bf16_t* xb  = Ab + elems;
  bf16_t* Wqb = xb + elems;
  bf16_t* Wkb = Wqb + W_ELEMS;
  bf16_t* Wvb = Wkb + W_ELEMS;
  bf16_t* Wob = Wvb + W_ELEMS;
  float* rcp = (float*)(Wob + W_ELEMS);  // NBH*SEQ floats

  // one-shot fp32 -> bf16 conversions (bandwidth-trivial)
  hipLaunchKernelGGL(cvt_bf16_kernel, dim3((int)(elems / 2048)), dim3(256), 0, stream,
                     x, xb, (int)elems);
  hipLaunchKernelGGL(cvt_bf16_kernel, dim3(W_ELEMS / 2048), dim3(256), 0, stream,
                     Wq, Wqb, W_ELEMS);
  hipLaunchKernelGGL(cvt_bf16_kernel, dim3(W_ELEMS / 2048), dim3(256), 0, stream,
                     Wk, Wkb, W_ELEMS);
  hipLaunchKernelGGL(cvt_bf16_kernel, dim3(W_ELEMS / 2048), dim3(256), 0, stream,
                     Wv, Wvb, W_ELEMS);
  hipLaunchKernelGGL(cvt_bf16_kernel, dim3(W_ELEMS / 2048), dim3(256), 0, stream,
                     Wo, Wob, W_ELEMS);

  hipLaunchKernelGGL(qkv_proj_kernel, dim3(NROWS / 16, 4), dim3(256), 0, stream,
                     xb, Wqb, Wkb, Wvb, Qb, Kb, Vb);
  hipLaunchKernelGGL(colsum_kernel, dim3(SEQ / 128, NBH), dim3(256), 0, stream,
                     Qb, Kb, rcp);
  hipLaunchKernelGGL(attn_out_kernel, dim3(SEQ / 128, NBH), dim3(256), 0, stream,
                     Qb, Kb, Vb, rcp, Ab);
  hipLaunchKernelGGL(out_proj_kernel, dim3(NROWS / 16, 4), dim3(256), 0, stream,
                     Ab, Wob, bo, out);
}
